// Head_36455682408801
// MI455X (gfx1250) — matrix operand order repaired and checked by reading, not compiled, not run on hardware
//
#include <hip/hip_runtime.h>

// ---------------------------------------------------------------------------
// Causal attention head, B=16, T=2048, E=H=128, fp32 in/out.
// bf16 WMMA (v_wmma_f32_16x16x32_bf16) for QKV projection + flash attention.
// ---------------------------------------------------------------------------

#define BATCH 16
#define SEQ   2048
#define EMB   128
#define HEAD  128

typedef __bf16 bf16_t;
typedef __attribute__((ext_vector_type(16))) __bf16 v16bf;
typedef __attribute__((ext_vector_type(8)))  __bf16 v8bf;
typedef __attribute__((ext_vector_type(8)))  float  v8f;
typedef __attribute__((ext_vector_type(4)))  float  v4f;

#define WMMA_BF16(A, B, C) \
    __builtin_amdgcn_wmma_f32_16x16x32_bf16(false, (A), false, (B), (short)0, (C), false, false)

static __device__ __forceinline__ v16bf cat8(v8bf lo, v8bf hi) {
    return __builtin_shufflevector(lo, hi, 0,1,2,3,4,5,6,7,8,9,10,11,12,13,14,15);
}

// ---------------------------------------------------------------------------
// Kernel 1: Q/K/V projection.  grid = (256, 3), block = 256 (8 waves).
//   mat 0: Qb[B,T,H] = x@Wq + bq  (bf16, row-major)
//   mat 1: Kb[B,T,H] = x@Wk + bk  (bf16, row-major)
//   mat 2: VTb[B,H,T] = (x@Wv + bv)^T  (bf16, transposed per batch)
// ---------------------------------------------------------------------------
__global__ __launch_bounds__(256) void qkv_proj_kernel(
    const float* __restrict__ x,
    const float* __restrict__ Wq, const float* __restrict__ bq,
    const float* __restrict__ Wk, const float* __restrict__ bk,
    const float* __restrict__ Wv, const float* __restrict__ bv,
    bf16_t* __restrict__ Qb, bf16_t* __restrict__ Kb, bf16_t* __restrict__ VTb)
{
    const int mat = blockIdx.y;
    const float* W    = (mat == 0) ? Wq : (mat == 1) ? Wk : Wv;
    const float* bias = (mat == 0) ? bq : (mat == 1) ? bk : bv;

    // Stage W^T into LDS as bf16: wt[h][e] = W[e][h], row stride 136 (pad).
    __shared__ bf16_t wt[128 * 136];
    for (int i = threadIdx.x; i < 128 * 128; i += 256) {
        int e = i >> 7, h = i & 127;             // contiguous global read over h
        wt[h * 136 + e] = (bf16_t)W[i];
    }
    __syncthreads();

    const int wave = threadIdx.x >> 5;
    const int lane = threadIdx.x & 31;
    const int L2 = lane & 15, hi = lane >> 4;

    const int rb = (blockIdx.x * 8 + wave) * 16;      // global row base (over B*T)

    // A fragments of x (16 rows x 32 e per chunk), fp32 -> bf16.
    v16bf afrag[4];
    const float* xrow = x + (size_t)(rb + L2) * EMB;
#pragma unroll
    for (int c = 0; c < 4; ++c) {
        v4f a0 = *(const v4f*)(xrow + c * 32 + hi * 8);
        v4f a1 = *(const v4f*)(xrow + c * 32 + hi * 8 + 4);
        v4f a2 = *(const v4f*)(xrow + c * 32 + 16 + hi * 8);
        v4f a3 = *(const v4f*)(xrow + c * 32 + 16 + hi * 8 + 4);
#pragma unroll
        for (int i = 0; i < 4; ++i) {
            afrag[c][i]      = (bf16_t)a0[i];
            afrag[c][4 + i]  = (bf16_t)a1[i];
            afrag[c][8 + i]  = (bf16_t)a2[i];
            afrag[c][12 + i] = (bf16_t)a3[i];
        }
    }

#pragma unroll
    for (int t = 0; t < 8; ++t) {
        const float bb = bias[t * 16 + L2];           // bias of this lane's column
        v8f acc;
#pragma unroll
        for (int i = 0; i < 8; ++i) acc[i] = bb;

#pragma unroll
        for (int c = 0; c < 4; ++c) {
            const bf16_t* wp = &wt[(t * 16 + L2) * 136 + c * 32 + hi * 8];
            v16bf bfrag = cat8(*(const v8bf*)wp, *(const v8bf*)(wp + 16));
            acc = WMMA_BF16(afrag[c], bfrag, acc);
        }

        if (mat < 2) {
            bf16_t* out = (mat == 0) ? Qb : Kb;       // [B*T, H] row-major
#pragma unroll
            for (int r = 0; r < 8; ++r)
                out[(size_t)(rb + hi * 8 + r) * HEAD + t * 16 + L2] = (bf16_t)acc[r];
        } else {
            const int bidx = rb >> 11;                // rb / SEQ
            const int tloc = rb & (SEQ - 1);          // rb % SEQ
            bf16_t* out = VTb + ((size_t)(bidx * HEAD + t * 16 + L2)) * SEQ + tloc + hi * 8;
#pragma unroll
            for (int r = 0; r < 8; ++r) out[r] = (bf16_t)acc[r];
        }
    }
}

// ---------------------------------------------------------------------------
// Kernel 2: causal flash attention.  grid = 256, block = 256 (8 waves).
// Each wave: one (batch, 16-query tile), full head dim.
// Computes S^T = K_tile @ Q^T and O^T += V^T @ P^T so that softmax rows are
// lane-resident and the P layout fixup is a single half-wave shuffle.
// ---------------------------------------------------------------------------
__global__ __launch_bounds__(256) void flash_attn_kernel(
    const bf16_t* __restrict__ Qb, const bf16_t* __restrict__ Kb,
    const bf16_t* __restrict__ VTb, float* __restrict__ out)
{
    const int wave = threadIdx.x >> 5;
    const int lane = threadIdx.x & 31;
    const int L2 = lane & 15, hi = lane >> 4;
    const bool lo = (hi == 0);

    const int wid = blockIdx.x * 8 + wave;            // 0..2047
    const int b  = wid >> 7;                          // batch
    const int qb = (wid & 127) * 16;                  // query tile base

    // Q^T B-fragments: lane holds query column qb+L2, e-chunk per fragment.
    v16bf qfrag[4];
    {
        const bf16_t* qrow = Qb + (size_t)(b * SEQ + qb + L2) * HEAD;
#pragma unroll
        for (int c = 0; c < 4; ++c) {
            const bf16_t* p = qrow + c * 32 + hi * 8;
            qfrag[c] = cat8(*(const v8bf*)p, *(const v8bf*)(p + 16));
        }
    }

    const bf16_t* kbase  = Kb  + (size_t)b * SEQ * HEAD;
    const bf16_t* vtbase = VTb + (size_t)b * HEAD * SEQ;
    const float scale = 0.08838834764831845f;         // 1/sqrt(128)

    float m = -3.0e38f, l = 0.0f;
    v8f o[8];
#pragma unroll
    for (int t = 0; t < 8; ++t)
#pragma unroll
        for (int i = 0; i < 8; ++i) o[t][i] = 0.0f;

    for (int j0 = 0; j0 < qb + 16; j0 += 32) {
        // ---- S^T tiles: keys j0..j0+15 (s0) and j0+16..j0+31 (s1) ----
        v8f s0, s1;
#pragma unroll
        for (int i = 0; i < 8; ++i) { s0[i] = 0.0f; s1[i] = 0.0f; }
#pragma unroll
        for (int c = 0; c < 4; ++c) {
            const bf16_t* kr0 = kbase + (size_t)(j0 + L2) * HEAD + c * 32;
            const bf16_t* kr1 = kbase + (size_t)(j0 + 16 + L2) * HEAD + c * 32;
            v16bf ka = cat8(*(const v8bf*)(kr0 + hi * 8), *(const v8bf*)(kr0 + 16 + hi * 8));
            v16bf kc = cat8(*(const v8bf*)(kr1 + hi * 8), *(const v8bf*)(kr1 + 16 + hi * 8));
            s0 = WMMA_BF16(ka, qfrag[c], s0);
            s1 = WMMA_BF16(kc, qfrag[c], s1);
        }

        // ---- scale + causal mask + online softmax (rows are lane-resident) ----
        const int q = qb + L2;
        float p0[8], p1[8];
        float mx = -3.0e38f;
#pragma unroll
        for (int r = 0; r < 8; ++r) {
            const int k0 = j0 + r + hi * 8;
            const float v0 = (k0 <= q)      ? s0[r] * scale : -1.0e30f;
            const float v1 = (k0 + 16 <= q) ? s1[r] * scale : -1.0e30f;
            p0[r] = v0; p1[r] = v1;
            mx = fmaxf(mx, fmaxf(v0, v1));
        }
        mx = fmaxf(mx, __shfl_xor(mx, 16, 32));       // combine lo/hi key halves
        const float mn = fmaxf(m, mx);
        const float alpha = __expf(m - mn);
        float rs = 0.0f;
#pragma unroll
        for (int r = 0; r < 8; ++r) {
            p0[r] = __expf(p0[r] - mn);
            p1[r] = __expf(p1[r] - mn);
            rs += p0[r] + p1[r];
        }
        rs += __shfl_xor(rs, 16, 32);
        l = l * alpha + rs;
        m = mn;
#pragma unroll
        for (int t = 0; t < 8; ++t)
#pragma unroll
            for (int i = 0; i < 8; ++i) o[t][i] *= alpha;

        // ---- P^T B-fragment: half-wave exchange, no LDS ----
        v16bf pfrag;
#pragma unroll
        for (int r = 0; r < 8; ++r) {
            const float xp0 = __shfl_xor(p0[r], 16, 32);
            const float xp1 = __shfl_xor(p1[r], 16, 32);
            pfrag[r]     = (bf16_t)(lo ? p0[r] : xp1);
            pfrag[r + 8] = (bf16_t)(lo ? xp0 : p1[r]);
        }

        // ---- O^T += V^T @ P^T ----
#pragma unroll
        for (int t = 0; t < 8; ++t) {
            const bf16_t* vr = vtbase + (size_t)(t * 16 + L2) * SEQ + j0;
            v16bf va = cat8(*(const v8bf*)(vr + hi * 16), *(const v8bf*)(vr + hi * 16 + 8));
            o[t] = WMMA_BF16(va, pfrag, o[t]);
        }
    }

    // ---- normalize + store: per lane 8 contiguous floats per tile ----
    const float linv = 1.0f / l;
    float* orow = out + (size_t)(b * SEQ + qb + L2) * HEAD;
#pragma unroll
    for (int t = 0; t < 8; ++t) {
        v4f w0, w1;
#pragma unroll
        for (int r = 0; r < 4; ++r) { w0[r] = o[t][r] * linv; w1[r] = o[t][4 + r] * linv; }
        *(v4f*)(orow + t * 16 + hi * 8)     = w0;
        *(v4f*)(orow + t * 16 + hi * 8 + 4) = w1;
    }
}

// ---------------------------------------------------------------------------
extern "C" void kernel_launch(void* const* d_in, const int* in_sizes, int n_in,
                              void* d_out, int out_size, void* d_ws, size_t ws_size,
                              hipStream_t stream) {
    const float* x  = (const float*)d_in[0];
    const float* Wq = (const float*)d_in[1];
    const float* bq = (const float*)d_in[2];
    const float* Wk = (const float*)d_in[3];
    const float* bk = (const float*)d_in[4];
    const float* Wv = (const float*)d_in[5];
    const float* bv = (const float*)d_in[6];

    const size_t elems = (size_t)BATCH * SEQ * HEAD;   // 4,194,304
    bf16_t* Qb  = (bf16_t*)d_ws;
    bf16_t* Kb  = Qb + elems;
    bf16_t* VTb = Kb + elems;                          // total 24 MB of ws

    // B*T/16 = 2048 wave tiles, 8 waves/block -> 256 blocks; y = {Q,K,V}
    qkv_proj_kernel<<<dim3(256, 3), 256, 0, stream>>>(
        x, Wq, bq, Wk, bk, Wv, bv, Qb, Kb, VTb);

    // 16 batches * 128 query tiles = 2048 waves -> 256 blocks
    flash_attn_kernel<<<dim3(256), 256, 0, stream>>>(Qb, Kb, VTb, (float*)d_out);
}
